// EncodeProcessDecode_14431090115093
// MI455X (gfx1250) — matrix operand order repaired and checked by reading, not compiled, not run on hardware
//
#include <hip/hip_runtime.h>

// ---------------------------------------------------------------------------
// Types for CDNA5 WMMA / TDM (gfx1250, wave32)
// ---------------------------------------------------------------------------
typedef __bf16 bf16;
typedef __attribute__((ext_vector_type(16))) __bf16 v16bf;
typedef __attribute__((ext_vector_type(8)))  float  v8f;
typedef __attribute__((ext_vector_type(4)))  unsigned u32x4;
typedef __attribute__((ext_vector_type(8)))  int      i32x8;
typedef __attribute__((ext_vector_type(4)))  int      i32x4;

#define F_IN   30
#define LATENT 96
#define HID    128
#define STEPS  10
#define OUTF   3

union AFragU { v16bf v; unsigned u[8]; };

// Order-preserving float <-> uint map for atomic max on floats (init = enc(-inf))
__device__ __forceinline__ unsigned f2ord(float f) {
    unsigned u = __float_as_uint(f);
    return (u & 0x80000000u) ? ~u : (u | 0x80000000u);
}
__device__ __forceinline__ float ord2f(unsigned u) {
    return (u & 0x80000000u) ? __uint_as_float(u & 0x7fffffffu)
                             : __uint_as_float(~u);
}
#define ORD_NEG_INF 0x007FFFFFu  // f2ord(-inf)

__device__ __forceinline__ float bf_lo(unsigned w) { return __uint_as_float(w << 16); }
__device__ __forceinline__ float bf_hi(unsigned w) { return __uint_as_float(w & 0xffff0000u); }

// Load a 16x32 bf16 A-fragment from a row-major LDS tile (stride in bf16 elems).
// ISA layout: lanes 0-15 (M=lane): V0..3 = K 0..7, V4..7 = K 16..23
//             lanes 16-31 (M=lane-16): V0..3 = K 8..15, V4..7 = K 24..31
__device__ __forceinline__ v16bf load_a_frag(const bf16* As, int stride, int lane, int k0) {
    const int m  = lane & 15;
    const int hh = (lane >> 4) << 3;  // 0 or 8
    const bf16* row = As + m * stride + k0 + hh;
    AFragU f;
#pragma unroll
    for (int v = 0; v < 4; ++v) {
        f.u[v]     = *(const unsigned*)(row + 2 * v);
        f.u[v + 4] = *(const unsigned*)(row + 16 + 2 * v);
    }
    return f.v;
}

__device__ __forceinline__ v8f wmma_bf16(v16bf a, v16bf b, v8f c) {
    return __builtin_amdgcn_wmma_f32_16x16x32_bf16(false, a, false, b,
                                                   (short)0, c, false, false);
}

// ---------------------------------------------------------------------------
// TDM: 2-D bf16 tile DMA global -> LDS (contiguous LDS rows, no padding).
//   gbase : global address of tile start (row-major, row stride == dim0 elems)
//   ldsAddr: LDS byte offset of destination
//   dim0  : row width in elements (== tensor_dim0 == tensor_dim0_stride)
//   rows  : tile_dim1
//   tensorRows: rows remaining in tensor from gbase (OOB guard)
// Descriptor per CDNA5 ISA ch.8 (D# group0/group1), groups 2/3 unused (zero).
// ---------------------------------------------------------------------------
__device__ __forceinline__ void tdm_load_2d_bf16(const bf16* gbase, unsigned ldsAddr,
                                                 int dim0, int rows, int tensorRows) {
    const unsigned long long ga = (unsigned long long)(size_t)gbase;
    u32x4 g0;
    g0.x = 1u;                                        // count=1, user mode
    g0.y = ldsAddr;                                   // lds_addr [63:32]
    g0.z = (unsigned)(ga & 0xffffffffu);              // global_addr [95:64]
    g0.w = (unsigned)((ga >> 32) & 0x01ffffffu) | (2u << 30);  // addr hi + type=2
    const unsigned d0 = (unsigned)dim0;
    const unsigned tr = (unsigned)tensorRows;
    i32x8 g1;
    g1[0] = (int)(1u << 16);                          // data_size = 2 bytes
    g1[1] = (int)((d0 & 0xffffu) << 16);              // tensor_dim0[15:0] -> [63:48]
    g1[2] = (int)((d0 >> 16) | ((tr & 0xffffu) << 16));   // tdim0 hi | tdim1 lo
    g1[3] = (int)((tr >> 16) | (d0 << 16));           // tdim1 hi | tile_dim0
    g1[4] = (int)(unsigned)rows;                      // tile_dim1 (tile_dim2=0)
    g1[5] = (int)d0;                                  // tensor_dim0_stride[31:0]
    g1[6] = 0;
    g1[7] = 0;
    const i32x4 z4 = {0, 0, 0, 0};
    const i32x8 z8 = {0, 0, 0, 0, 0, 0, 0, 0};
    __builtin_amdgcn_tensor_load_to_lds(g0, g1, z4, z4, z8, 0);
}

__device__ __forceinline__ unsigned lds_addr_of(const void* p) {
    return (unsigned)(size_t)p;   // low 32 bits of flat LDS address = LDS offset
}

// ---------------------------------------------------------------------------
// Weight pack: f32 row-major [K x NC] (optionally concat A;B along K, zero pad)
// -> bf16 in WMMA B-fragment order:
//   dst[((kc*CT+ct)*32 + lane)*16 + j] <= W[kc*32 + 8*(lane>>4) + (j&7) + 16*(j>>3), ct*16 + (lane&15)]
// ---------------------------------------------------------------------------
__global__ void k_pack_weight(bf16* __restrict__ dst,
                              const float* __restrict__ srcA,
                              const float* __restrict__ srcB,
                              int K, int NC, int Ksplit, int srcNC, int validN) {
    const int total = K * NC;
    const int CT = NC >> 4;
    for (int idx = blockIdx.x * blockDim.x + threadIdx.x; idx < total;
         idx += gridDim.x * blockDim.x) {
        const int j    = idx & 15;
        const int lane = (idx >> 4) & 31;
        const int t    = idx >> 9;
        const int ct   = t % CT;
        const int kc   = t / CT;
        const int k = kc * 32 + ((lane >> 4) << 3) + (j & 7) + ((j >> 3) << 4);
        const int n = ct * 16 + (lane & 15);
        float v = 0.0f;
        if (n < validN && k < K) {
            if (k < Ksplit)      v = srcA[(long)k * srcNC + n];
            else if (srcB)       v = srcB[(long)(k - Ksplit) * srcNC + n];
        }
        dst[idx] = (bf16)v;
    }
}

// ---------------------------------------------------------------------------
// Encoder: x[N,30] -> relu(W1) -> relu(W2) -> W3 -> LayerNorm -> h(f32) + h_bf
// ---------------------------------------------------------------------------
__global__ void __launch_bounds__(256)
k_encoder(const float* __restrict__ x,
          const bf16* __restrict__ W1, const bf16* __restrict__ W2,
          const bf16* __restrict__ W3,
          const float* __restrict__ b1, const float* __restrict__ b2,
          const float* __restrict__ b3,
          const float* __restrict__ lng, const float* __restrict__ lnb,
          float* __restrict__ h, bf16* __restrict__ hb, int Nn) {
    __shared__ __align__(16) bf16 sA[16 * 136];
    __shared__ __align__(16) bf16 sB[16 * 136];
    __shared__ float sO[16 * 96];
    const int tid = threadIdx.x, lane = tid & 31, wave = tid >> 5;
    const int row0 = blockIdx.x * 16;
    const int hf = lane >> 4;

    for (int idx = tid; idx < 16 * 32; idx += 256) {
        const int r = idx >> 5, c = idx & 31;
        const int gr = row0 + r;
        float v = 0.0f;
        if (c < F_IN && gr < Nn) v = x[gr * F_IN + c];
        sA[r * 136 + c] = (bf16)v;
    }
    __syncthreads();

    // GEMM1: [16x32] x [32x128] + b1, relu -> sB
    {
        const int ct = wave;
        v8f acc = {};
        v16bf a = load_a_frag(sA, 136, lane, 0);
        v16bf b = *(const v16bf*)(W1 + (ct * 32 + lane) * 16);
        acc = wmma_bf16(a, b, acc);
        const int n = ct * 16 + (lane & 15);
        const float bias = b1[n];
#pragma unroll
        for (int r = 0; r < 8; ++r) {
            float v = acc[r] + bias; v = v > 0.0f ? v : 0.0f;
            sB[(r + 8 * hf) * 136 + n] = (bf16)v;
        }
    }
    __syncthreads();

    // GEMM2: [16x128] x [128x128] + b2, relu -> sA
    {
        const int ct = wave;
        v8f acc = {};
#pragma unroll
        for (int kc = 0; kc < 4; ++kc) {
            v16bf a = load_a_frag(sB, 136, lane, kc * 32);
            v16bf b = *(const v16bf*)(W2 + ((kc * 8 + ct) * 32 + lane) * 16);
            acc = wmma_bf16(a, b, acc);
        }
        const int n = ct * 16 + (lane & 15);
        const float bias = b2[n];
#pragma unroll
        for (int r = 0; r < 8; ++r) {
            float v = acc[r] + bias; v = v > 0.0f ? v : 0.0f;
            sA[(r + 8 * hf) * 136 + n] = (bf16)v;
        }
    }
    __syncthreads();

    // GEMM3: [16x128] x [128x96] + b3 -> sO (f32)
    if (wave < 6) {
        const int ct = wave;
        v8f acc = {};
#pragma unroll
        for (int kc = 0; kc < 4; ++kc) {
            v16bf a = load_a_frag(sA, 136, lane, kc * 32);
            v16bf b = *(const v16bf*)(W3 + ((kc * 6 + ct) * 32 + lane) * 16);
            acc = wmma_bf16(a, b, acc);
        }
        const int n = ct * 16 + (lane & 15);
        const float bias = b3[n];
#pragma unroll
        for (int r = 0; r < 8; ++r) sO[(r + 8 * hf) * 96 + n] = acc[r] + bias;
    }
    __syncthreads();

    if (tid < 16) {
        const int gr = row0 + tid;
        if (gr < Nn) {
            float mu = 0.0f;
            for (int c = 0; c < 96; ++c) mu += sO[tid * 96 + c];
            mu *= (1.0f / 96.0f);
            float var = 0.0f;
            for (int c = 0; c < 96; ++c) {
                const float d = sO[tid * 96 + c] - mu;
                var += d * d;
            }
            var *= (1.0f / 96.0f);
            const float rs = rsqrtf(var + 1e-5f);
            for (int c = 0; c < 96; ++c) {
                const float v = (sO[tid * 96 + c] - mu) * rs * lng[c] + lnb[c];
                h[(long)gr * 96 + c]  = v;
                hb[(long)gr * 96 + c] = (bf16)v;
            }
        }
    }
}

// ---------------------------------------------------------------------------
__global__ void k_init_agg(unsigned* __restrict__ agg, long total) {
    long i = (long)blockIdx.x * blockDim.x + threadIdx.x;
    const long st = (long)gridDim.x * blockDim.x;
    for (; i < total; i += st) agg[i] = ORD_NEG_INF;
}

// ---------------------------------------------------------------------------
// Edge scatter on bf16 features: agg[dst] = max(agg[dst], hb[src]); 8 feats/thread.
// grid: (edge blocks, F/8)
// ---------------------------------------------------------------------------
__global__ void __launch_bounds__(256)
k_edge_max(const bf16* __restrict__ hb, int F,
           const int* __restrict__ src, const int* __restrict__ dst,
           unsigned* __restrict__ agg, int E) {
    const int e = blockIdx.x * blockDim.x + threadIdx.x;
    if (e >= E) return;
    const int c8 = blockIdx.y << 3;
    const int s = src[e], d = dst[e];
    const uint4 v = *(const uint4*)(hb + (long)s * F + c8);
    unsigned* a = agg + (long)d * F + c8;
    atomicMax(a + 0, f2ord(bf_lo(v.x)));
    atomicMax(a + 1, f2ord(bf_hi(v.x)));
    atomicMax(a + 2, f2ord(bf_lo(v.y)));
    atomicMax(a + 3, f2ord(bf_hi(v.y)));
    atomicMax(a + 4, f2ord(bf_lo(v.z)));
    atomicMax(a + 5, f2ord(bf_hi(v.z)));
    atomicMax(a + 6, f2ord(bf_lo(v.w)));
    atomicMax(a + 7, f2ord(bf_hi(v.w)));
}

// ---------------------------------------------------------------------------
// SAGE1: h1b = relu([finite(agg96) | h] @ [W1l;W1r] + b1l)   (K=192 -> N=128)
// agg half decoded by VALU into sAgg; h half DMA'd by TDM into sH.
// ---------------------------------------------------------------------------
__global__ void __launch_bounds__(256)
k_sage1(const unsigned* __restrict__ agg, const bf16* __restrict__ hb,
        const bf16* __restrict__ Wc, const float* __restrict__ bias,
        bf16* __restrict__ h1b, int Nn) {
    __shared__ __align__(16) bf16 sAgg[16 * 104];
    __shared__ __align__(16) bf16 sH[16 * 96];
    const int tid = threadIdx.x, lane = tid & 31, wave = tid >> 5;
    const int row0 = blockIdx.x * 16;
    const int hf = lane >> 4;

    if (wave == 0)   // kick the DMA for the dense h panel
        tdm_load_2d_bf16(hb + (long)row0 * 96, lds_addr_of(sH), 96, 16, Nn - row0);

    for (int idx = tid; idx < 16 * 96; idx += 256) {
        const int r = idx / 96, c = idx - r * 96;
        const int gr = row0 + r;
        float v = 0.0f;
        if (gr < Nn) {
            const float f = ord2f(agg[(long)gr * 96 + c]);
            v = (f > -3.0e38f && f < 3.0e38f) ? f : 0.0f;  // where(isfinite,.,0)
        }
        sAgg[r * 104 + c] = (bf16)v;
    }
    if (wave == 0) __builtin_amdgcn_s_wait_tensorcnt(0);
    __syncthreads();

    const int ct = wave;  // 8 col tiles of 128
    v8f acc = {};
#pragma unroll
    for (int kc = 0; kc < 3; ++kc) {
        v16bf a = load_a_frag(sAgg, 104, lane, kc * 32);
        v16bf b = *(const v16bf*)(Wc + ((kc * 8 + ct) * 32 + lane) * 16);
        acc = wmma_bf16(a, b, acc);
    }
#pragma unroll
    for (int kc = 3; kc < 6; ++kc) {
        v16bf a = load_a_frag(sH, 96, lane, (kc - 3) * 32);
        v16bf b = *(const v16bf*)(Wc + ((kc * 8 + ct) * 32 + lane) * 16);
        acc = wmma_bf16(a, b, acc);
    }
    const int n = ct * 16 + (lane & 15);
    const float bb = bias[n];
#pragma unroll
    for (int r = 0; r < 8; ++r) {
        const int gm = row0 + r + 8 * hf;
        if (gm < Nn) {
            float v = acc[r] + bb; v = v > 0.0f ? v : 0.0f;
            h1b[(long)gm * 128 + n] = (bf16)v;
        }
    }
}

// ---------------------------------------------------------------------------
// SAGE2 + residual + LayerNorm:
//   h = LN( [finite(agg128) | h1b] @ [W2l;W2r] + b2l + h );  also refresh h_bf.
// ---------------------------------------------------------------------------
__global__ void __launch_bounds__(256)
k_sage2(const unsigned* __restrict__ agg, const bf16* __restrict__ h1b,
        float* __restrict__ h, bf16* __restrict__ hb,
        const bf16* __restrict__ Wc, const float* __restrict__ bias,
        const float* __restrict__ lng, const float* __restrict__ lnb, int Nn) {
    __shared__ __align__(16) bf16 sAgg[16 * 136];
    __shared__ __align__(16) bf16 sH1[16 * 128];
    __shared__ float sO[16 * 96];
    const int tid = threadIdx.x, lane = tid & 31, wave = tid >> 5;
    const int row0 = blockIdx.x * 16;
    const int hf = lane >> 4;

    if (wave == 0)
        tdm_load_2d_bf16(h1b + (long)row0 * 128, lds_addr_of(sH1), 128, 16, Nn - row0);

    for (int idx = tid; idx < 16 * 128; idx += 256) {
        const int r = idx >> 7, c = idx & 127;
        const int gr = row0 + r;
        float v = 0.0f;
        if (gr < Nn) {
            const float f = ord2f(agg[(long)gr * 128 + c]);
            v = (f > -3.0e38f && f < 3.0e38f) ? f : 0.0f;
        }
        sAgg[r * 136 + c] = (bf16)v;
    }
    if (wave == 0) __builtin_amdgcn_s_wait_tensorcnt(0);
    __syncthreads();

    if (wave < 6) {
        const int ct = wave;  // 6 col tiles of 96
        v8f acc = {};
#pragma unroll
        for (int kc = 0; kc < 4; ++kc) {
            v16bf a = load_a_frag(sAgg, 136, lane, kc * 32);
            v16bf b = *(const v16bf*)(Wc + ((kc * 6 + ct) * 32 + lane) * 16);
            acc = wmma_bf16(a, b, acc);
        }
#pragma unroll
        for (int kc = 4; kc < 8; ++kc) {
            v16bf a = load_a_frag(sH1, 128, lane, (kc - 4) * 32);
            v16bf b = *(const v16bf*)(Wc + ((kc * 6 + ct) * 32 + lane) * 16);
            acc = wmma_bf16(a, b, acc);
        }
        const int n = ct * 16 + (lane & 15);
        const float bb = bias[n];
#pragma unroll
        for (int r = 0; r < 8; ++r) {
            const int gm = row0 + r + 8 * hf;
            const float res = (gm < Nn) ? h[(long)gm * 96 + n] : 0.0f;
            sO[(r + 8 * hf) * 96 + n] = acc[r] + bb + res;
        }
    }
    __syncthreads();

    if (tid < 16) {
        const int gr = row0 + tid;
        if (gr < Nn) {
            float mu = 0.0f;
            for (int c = 0; c < 96; ++c) mu += sO[tid * 96 + c];
            mu *= (1.0f / 96.0f);
            float var = 0.0f;
            for (int c = 0; c < 96; ++c) {
                const float d = sO[tid * 96 + c] - mu;
                var += d * d;
            }
            var *= (1.0f / 96.0f);
            const float rs = rsqrtf(var + 1e-5f);
            for (int c = 0; c < 96; ++c) {
                const float v = (sO[tid * 96 + c] - mu) * rs * lng[c] + lnb[c];
                h[(long)gr * 96 + c]  = v;
                hb[(long)gr * 96 + c] = (bf16)v;
            }
        }
    }
}

// ---------------------------------------------------------------------------
// Decoder: h_bf[N,96] -> relu(W1) -> relu(W2) -> W3[128,3] -> out[N,3]
// Input tile DMA'd via TDM.
// ---------------------------------------------------------------------------
__global__ void __launch_bounds__(256)
k_decoder(const bf16* __restrict__ hb, const bf16* __restrict__ W1,
          const bf16* __restrict__ W2, const bf16* __restrict__ W3,
          const float* __restrict__ b1, const float* __restrict__ b2,
          const float* __restrict__ b3, float* __restrict__ out, int Nn) {
    __shared__ __align__(16) bf16 sH[16 * 96];
    __shared__ __align__(16) bf16 sB1[16 * 136];
    __shared__ __align__(16) bf16 sB2[16 * 136];
    const int tid = threadIdx.x, lane = tid & 31, wave = tid >> 5;
    const int row0 = blockIdx.x * 16;
    const int hf = lane >> 4;

    if (wave == 0) {
        tdm_load_2d_bf16(hb + (long)row0 * 96, lds_addr_of(sH), 96, 16, Nn - row0);
        __builtin_amdgcn_s_wait_tensorcnt(0);
    }
    __syncthreads();

    // GEMM1: [16x96] x [96x128] + b1, relu -> sB1
    {
        const int ct = wave;
        v8f acc = {};
#pragma unroll
        for (int kc = 0; kc < 3; ++kc) {
            v16bf a = load_a_frag(sH, 96, lane, kc * 32);
            v16bf b = *(const v16bf*)(W1 + ((kc * 8 + ct) * 32 + lane) * 16);
            acc = wmma_bf16(a, b, acc);
        }
        const int n = ct * 16 + (lane & 15);
        const float bias = b1[n];
#pragma unroll
        for (int r = 0; r < 8; ++r) {
            float v = acc[r] + bias; v = v > 0.0f ? v : 0.0f;
            sB1[(r + 8 * hf) * 136 + n] = (bf16)v;
        }
    }
    __syncthreads();

    // GEMM2: [16x128] x [128x128] + b2, relu -> sB2
    {
        const int ct = wave;
        v8f acc = {};
#pragma unroll
        for (int kc = 0; kc < 4; ++kc) {
            v16bf a = load_a_frag(sB1, 136, lane, kc * 32);
            v16bf b = *(const v16bf*)(W2 + ((kc * 8 + ct) * 32 + lane) * 16);
            acc = wmma_bf16(a, b, acc);
        }
        const int n = ct * 16 + (lane & 15);
        const float bias = b2[n];
#pragma unroll
        for (int r = 0; r < 8; ++r) {
            float v = acc[r] + bias; v = v > 0.0f ? v : 0.0f;
            sB2[(r + 8 * hf) * 136 + n] = (bf16)v;
        }
    }
    __syncthreads();

    // GEMM3: [16x128] x [128x16(3 valid)] + b3 -> out
    if (wave == 0) {
        v8f acc = {};
#pragma unroll
        for (int kc = 0; kc < 4; ++kc) {
            v16bf a = load_a_frag(sB2, 136, lane, kc * 32);
            v16bf b = *(const v16bf*)(W3 + (kc * 32 + lane) * 16);
            acc = wmma_bf16(a, b, acc);
        }
        const int n = lane & 15;
        if (n < OUTF) {
            const float bias = b3[n];
#pragma unroll
            for (int r = 0; r < 8; ++r) {
                const int gm = row0 + r + 8 * hf;
                if (gm < Nn) out[(long)gm * OUTF + n] = acc[r] + bias;
            }
        }
    }
}

// ---------------------------------------------------------------------------
// Host orchestration
// ---------------------------------------------------------------------------
static inline char* ws_take(char*& p, size_t bytes) {
    char* r = p;
    p += (bytes + 255) & ~(size_t)255;
    return r;
}

extern "C" void kernel_launch(void* const* d_in, const int* in_sizes, int n_in,
                              void* d_out, int out_size, void* d_ws, size_t ws_size,
                              hipStream_t stream) {
    const float* x        = (const float*)d_in[0];
    const int*   ei       = (const int*)d_in[1];
    const float* enc_W1   = (const float*)d_in[2];
    const float* enc_b1   = (const float*)d_in[3];
    const float* enc_W2   = (const float*)d_in[4];
    const float* enc_b2   = (const float*)d_in[5];
    const float* enc_W3   = (const float*)d_in[6];
    const float* enc_b3   = (const float*)d_in[7];
    const float* enc_ln_g = (const float*)d_in[8];
    const float* enc_ln_b = (const float*)d_in[9];
    const float* p_W1l    = (const float*)d_in[10];
    const float* p_b1l    = (const float*)d_in[11];
    const float* p_W1r    = (const float*)d_in[12];
    const float* p_W2l    = (const float*)d_in[13];
    const float* p_b2l    = (const float*)d_in[14];
    const float* p_W2r    = (const float*)d_in[15];
    const float* p_ln_g   = (const float*)d_in[16];
    const float* p_ln_b   = (const float*)d_in[17];
    const float* dec_W1   = (const float*)d_in[18];
    const float* dec_b1   = (const float*)d_in[19];
    const float* dec_W2   = (const float*)d_in[20];
    const float* dec_b2   = (const float*)d_in[21];
    const float* dec_W3   = (const float*)d_in[22];
    const float* dec_b3   = (const float*)d_in[23];
    float* out = (float*)d_out;

    const int Nn = in_sizes[0] / F_IN;
    const int Ee = in_sizes[1] / 2;
    const int* src = ei;
    const int* dst = ei + Ee;

    // workspace partition
    char* p = (char*)d_ws;
    float*    h    = (float*)ws_take(p, (size_t)Nn * LATENT * sizeof(float));
    bf16*     hbf  = (bf16*)ws_take(p, (size_t)Nn * LATENT * sizeof(bf16));
    bf16*     h1b  = (bf16*)ws_take(p, (size_t)Nn * HID * sizeof(bf16));
    unsigned* agg  = (unsigned*)ws_take(p, (size_t)Nn * HID * sizeof(unsigned));
    bf16* ew1 = (bf16*)ws_take(p, (size_t)32 * 128 * sizeof(bf16));
    bf16* ew2 = (bf16*)ws_take(p, (size_t)128 * 128 * sizeof(bf16));
    bf16* ew3 = (bf16*)ws_take(p, (size_t)128 * 96 * sizeof(bf16));
    bf16* dw1 = (bf16*)ws_take(p, (size_t)96 * 128 * sizeof(bf16));
    bf16* dw2 = (bf16*)ws_take(p, (size_t)128 * 128 * sizeof(bf16));
    bf16* dw3 = (bf16*)ws_take(p, (size_t)128 * 16 * sizeof(bf16));
    bf16* wc1 = (bf16*)ws_take(p, (size_t)STEPS * 192 * 128 * sizeof(bf16));
    bf16* wc2 = (bf16*)ws_take(p, (size_t)STEPS * 256 * 96 * sizeof(bf16));
    (void)ws_size; (void)n_in; (void)out_size;

    // ---- pack all weights to bf16 WMMA-fragment layout ----
    auto packGrid = [](int K, int NC) { return dim3((unsigned)((K * NC + 255) / 256)); };
    k_pack_weight<<<packGrid(32, 128), 256, 0, stream>>>(ew1, enc_W1, nullptr, 32, 128, 30, 128, 128);
    k_pack_weight<<<packGrid(128, 128), 256, 0, stream>>>(ew2, enc_W2, nullptr, 128, 128, 128, 128, 128);
    k_pack_weight<<<packGrid(128, 96), 256, 0, stream>>>(ew3, enc_W3, nullptr, 128, 96, 128, 96, 96);
    k_pack_weight<<<packGrid(96, 128), 256, 0, stream>>>(dw1, dec_W1, nullptr, 96, 128, 96, 128, 128);
    k_pack_weight<<<packGrid(128, 128), 256, 0, stream>>>(dw2, dec_W2, nullptr, 128, 128, 128, 128, 128);
    k_pack_weight<<<packGrid(128, 16), 256, 0, stream>>>(dw3, dec_W3, nullptr, 128, 16, 128, 3, 3);
    for (int s = 0; s < STEPS; ++s) {
        k_pack_weight<<<packGrid(192, 128), 256, 0, stream>>>(
            wc1 + (size_t)s * 192 * 128,
            p_W1l + (size_t)s * 96 * 128, p_W1r + (size_t)s * 96 * 128,
            192, 128, 96, 128, 128);
        k_pack_weight<<<packGrid(256, 96), 256, 0, stream>>>(
            wc2 + (size_t)s * 256 * 96,
            p_W2l + (size_t)s * 128 * 96, p_W2r + (size_t)s * 128 * 96,
            256, 96, 128, 96, 96);
    }

    const int tiles = (Nn + 15) / 16;
    const dim3 egrid96((unsigned)((Ee + 255) / 256), 96 / 8);
    const dim3 egrid128((unsigned)((Ee + 255) / 256), 128 / 8);

    // ---- encoder ----
    k_encoder<<<tiles, 256, 0, stream>>>(x, ew1, ew2, ew3, enc_b1, enc_b2, enc_b3,
                                         enc_ln_g, enc_ln_b, h, hbf, Nn);

    // ---- message-passing steps ----
    for (int s = 0; s < STEPS; ++s) {
        {   // agg = segment_max(hbf[src] -> dst), F=96
            long tot = (long)Nn * 96;
            int blocks = (int)((tot + 255) / 256);
            k_init_agg<<<blocks, 256, 0, stream>>>(agg, tot);
            k_edge_max<<<egrid96, 256, 0, stream>>>(hbf, 96, src, dst, agg, Ee);
        }
        k_sage1<<<tiles, 256, 0, stream>>>(agg, hbf, wc1 + (size_t)s * 192 * 128,
                                           p_b1l + (size_t)s * 128, h1b, Nn);
        {   // agg = segment_max(h1b[src] -> dst), F=128
            long tot = (long)Nn * 128;
            int blocks = (int)((tot + 255) / 256);
            k_init_agg<<<blocks, 256, 0, stream>>>(agg, tot);
            k_edge_max<<<egrid128, 256, 0, stream>>>(h1b, 128, src, dst, agg, Ee);
        }
        k_sage2<<<tiles, 256, 0, stream>>>(agg, h1b, h, hbf,
                                           wc2 + (size_t)s * 256 * 96,
                                           p_b2l + (size_t)s * 96,
                                           p_ln_g + (size_t)s * 96,
                                           p_ln_b + (size_t)s * 96, Nn);
    }

    // ---- decoder ----
    k_decoder<<<tiles, 256, 0, stream>>>(hbf, dw1, dw2, dw3, dec_b1, dec_b2, dec_b3,
                                         out, Nn);
}
